// SelfAttention_87376814670512
// MI455X (gfx1250) — matrix operand order repaired and checked by reading, not compiled, not run on hardware
//
#include <hip/hip_runtime.h>

// ---------------------------------------------------------------------------
// Problem constants (from reference): B=4, L=2048, HID=2048, H=16, D=128
// ---------------------------------------------------------------------------
#define BB    4
#define LL    2048
#define HID   2048
#define NH    16
#define DD    128
#define BH    (BB * NH)          // 64
#define MROWS (BB * LL)          // 8192
#define NQKV  (3 * NH * DD)      // 6144

typedef __attribute__((ext_vector_type(16))) __bf16 v16bf;
typedef __attribute__((ext_vector_type(8)))  float  v8f;
typedef __attribute__((ext_vector_type(4)))  unsigned int v4u;
typedef __attribute__((ext_vector_type(8)))  int    v8i;

union Frag16 {
    v16bf v;
    uint4 q[2];
    unsigned short u[16];
};

static __device__ inline unsigned short f32_to_bf16(float f) {
    union { float f; unsigned int u; } a; a.f = f;
    unsigned int u = a.u;
    unsigned int r = u + 0x7FFFu + ((u >> 16) & 1u);   // round-to-nearest-even
    return (unsigned short)(r >> 16);
}
static __device__ inline float bf16_to_f32(unsigned short s) {
    union { unsigned int u; float f; } a; a.u = ((unsigned int)s) << 16;
    return a.f;
}
static __device__ inline v8f wmma_bf16(v16bf a, v16bf b, v8f c) {
    return __builtin_amdgcn_wmma_f32_16x16x32_bf16(false, a, false, b,
                                                   (short)0, c, false, false);
}
static __device__ inline void wave_fence() {
#if __has_builtin(__builtin_amdgcn_wave_barrier)
    __builtin_amdgcn_wave_barrier();
#else
    asm volatile("" ::: "memory");
#endif
}
template <int N>
static __device__ inline void wait_tensorcnt() {
#if __has_builtin(__builtin_amdgcn_s_wait_tensorcnt)
    __builtin_amdgcn_s_wait_tensorcnt(N);
#else
    asm volatile("s_wait_tensorcnt %0" :: "i"(N) : "memory");
#endif
}
// low 32 bits of a generic shared-memory pointer == LDS byte offset (ISA 10.2)
static __device__ inline unsigned lds_off(const void* p) {
    return (unsigned)(size_t)p;
}

// ---------------------------------------------------------------------------
// TDM: 2D tile load (16-bit elements) via Tensor Data Mover.
// Descriptor layout per CDNA5 ISA. Issue once per wave
// (EXEC ignored); completion tracked on TENSORcnt (in-order per wave).
// Element units: dims/strides in elements (data_size = 2 bytes).
// ---------------------------------------------------------------------------
static __device__ inline void tdm_load_2d_bf16(unsigned lds_byte_off,
                                               const void* gptr,
                                               unsigned tensor_d0, unsigned tensor_d1,
                                               unsigned tile_d0,  unsigned tile_d1,
                                               unsigned long long stride0) {
    unsigned long long ga = (unsigned long long)(size_t)gptr;
    v4u g0;
    g0[0] = 1u;                                           // count=1 (valid), user mode
    g0[1] = lds_byte_off;                                 // lds_addr
    g0[2] = (unsigned)(ga & 0xFFFFFFFFull);               // global_addr[31:0]
    g0[3] = (unsigned)((ga >> 32) & 0x1FFFFFFull)         // global_addr[56:32]
          | (2u << 30);                                   // type = 2 ("image")
    v8i g1;
    g1[0] = (int)(1u << 16);                              // data_size=1 -> 2 bytes
    g1[1] = (int)((tensor_d0 & 0xFFFFu) << 16);           // tensor_dim0[15:0]
    g1[2] = (int)((tensor_d0 >> 16) | ((tensor_d1 & 0xFFFFu) << 16));
    g1[3] = (int)((tensor_d1 >> 16) | (tile_d0 << 16));   // tile_dim0
    g1[4] = (int)(tile_d1 & 0xFFFFu);                     // tile_dim1 (tile_dim2=0)
    g1[5] = (int)(unsigned)(stride0 & 0xFFFFFFFFull);     // tensor_dim0_stride[31:0]
    g1[6] = (int)(unsigned)((stride0 >> 32) & 0xFFFFull); // stride[47:32]
    g1[7] = 0;
    asm volatile("tensor_load_to_lds %0, %1"
                 :: "s"(g0), "s"(g1)
                 : "memory");
}

// ---------------------------------------------------------------------------
// Kernel 1: QKV projection.  C[8192,6144] = X[8192,2048] @ Wqkv[2048,6144]
// Epilogue scatters bf16 into Q[BH,L,D], K[BH,L,D], Vt[BH,D,L].
// ---------------------------------------------------------------------------
__global__ __launch_bounds__(256)
void qkv_gemm_kernel(const float* __restrict__ X, const float* __restrict__ W,
                     unsigned short* __restrict__ qb,
                     unsigned short* __restrict__ kbuf,
                     unsigned short* __restrict__ vtb) {
    __shared__ unsigned short As[128 * 32];   // [m][k] bf16
    __shared__ unsigned short Bs[128 * 32];   // transposed: [n][k] bf16

    const int tid  = threadIdx.x;
    const int lane = tid & 31;
    const int w    = tid >> 5;
    const int wm   = w >> 1;
    const int wn   = w & 1;
    const int h    = lane >> 4;
    const int l16  = lane & 15;

    const int n0 = blockIdx.x * 128;
    const int m0 = blockIdx.y * 128;

    v8f acc[2][4];
    const v8f vzero = {0.f, 0.f, 0.f, 0.f, 0.f, 0.f, 0.f, 0.f};
#pragma unroll
    for (int a = 0; a < 2; ++a)
#pragma unroll
        for (int b = 0; b < 4; ++b) acc[a][b] = vzero;

    for (int k0 = 0; k0 < HID; k0 += 32) {
#pragma unroll
        for (int r = 0; r < 4; ++r) {
            int id  = tid + 256 * r;
            int row = id >> 3;
            int c4  = id & 7;
            float4 xv = *(const float4*)(X + (size_t)(m0 + row) * HID + k0 + c4 * 4);
            unsigned short* d = &As[row * 32 + c4 * 4];
            d[0] = f32_to_bf16(xv.x); d[1] = f32_to_bf16(xv.y);
            d[2] = f32_to_bf16(xv.z); d[3] = f32_to_bf16(xv.w);
        }
#pragma unroll
        for (int r = 0; r < 4; ++r) {
            int id  = tid + 256 * r;
            int row = id >> 5;
            int c4  = id & 31;
            float4 wv = *(const float4*)(W + (size_t)(k0 + row) * NQKV + n0 + c4 * 4);
            int n = c4 * 4;
            Bs[(n + 0) * 32 + row] = f32_to_bf16(wv.x);
            Bs[(n + 1) * 32 + row] = f32_to_bf16(wv.y);
            Bs[(n + 2) * 32 + row] = f32_to_bf16(wv.z);
            Bs[(n + 3) * 32 + row] = f32_to_bf16(wv.w);
        }
        __syncthreads();

        Frag16 afr[2];
#pragma unroll
        for (int ms = 0; ms < 2; ++ms) {
            int row = wm * 32 + ms * 16 + l16;
            afr[ms].q[0] = *(const uint4*)&As[row * 32 + 16 * h];
            afr[ms].q[1] = *(const uint4*)&As[row * 32 + 16 * h + 8];
        }
#pragma unroll
        for (int ns = 0; ns < 4; ++ns) {
            Frag16 bfr;
            int n = wn * 64 + ns * 16 + l16;
            bfr.q[0] = *(const uint4*)&Bs[n * 32 + 16 * h];
            bfr.q[1] = *(const uint4*)&Bs[n * 32 + 16 * h + 8];
#pragma unroll
            for (int ms = 0; ms < 2; ++ms)
                acc[ms][ns] = wmma_bf16(afr[ms].v, bfr.v, acc[ms][ns]);
        }
        __syncthreads();
    }

    const int t    = n0 / HID;                 // 0=Q 1=K 2=V
    const int head = (n0 % HID) / DD;
#pragma unroll
    for (int ms = 0; ms < 2; ++ms) {
#pragma unroll
        for (int ns = 0; ns < 4; ++ns) {
#pragma unroll
            for (int v = 0; v < 8; ++v) {
                float val = acc[ms][ns][v];
                int gm   = m0 + wm * 32 + ms * 16 + v + 8 * h;
                int b    = gm >> 11;
                int lpos = gm & 2047;
                int d    = wn * 64 + ns * 16 + l16;
                int bh   = b * NH + head;
                unsigned short bv = f32_to_bf16(val);
                if (t == 0)
                    qb[((size_t)bh * LL + lpos) * DD + d] = bv;
                else if (t == 1)
                    kbuf[((size_t)bh * LL + lpos) * DD + d] = bv;
                else
                    vtb[((size_t)bh * DD + d) * LL + lpos] = bv;
            }
        }
    }
}

// ---------------------------------------------------------------------------
// Kernel 2: RoPE in place on Q and K (bf16 storage, fp32 math).
// ---------------------------------------------------------------------------
__global__ __launch_bounds__(256)
void rope_kernel(unsigned short* __restrict__ qb, unsigned short* __restrict__ kbuf) {
    int gid = blockIdx.x * blockDim.x + threadIdx.x;
    int i   = gid & 63;
    int l   = (gid >> 6) & 2047;
    int bh  = gid >> 17;
    if (bh >= BH) return;

    float freq = __powf(10000.0f, -(float)i * (1.0f / 64.0f));
    float ang  = (float)l * freq;
    float s, c;
    __sincosf(ang, &s, &c);

    size_t base = ((size_t)bh * LL + l) * DD;

    float q1 = bf16_to_f32(qb[base + i]);
    float q2 = bf16_to_f32(qb[base + i + 64]);
    qb[base + i]      = f32_to_bf16(q1 * c - q2 * s);
    qb[base + i + 64] = f32_to_bf16(q2 * c + q1 * s);

    float k1 = bf16_to_f32(kbuf[base + i]);
    float k2 = bf16_to_f32(kbuf[base + i + 64]);
    kbuf[base + i]      = f32_to_bf16(k1 * c - k2 * s);
    kbuf[base + i + 64] = f32_to_bf16(k2 * c + k1 * s);
}

// ---------------------------------------------------------------------------
// Kernel 3: causal flash attention, double-buffered TDM K/V staging.
// Block = 4 waves, 64 Q rows; wave w owns rows qblock + w*16.
// Tile i+1 is DMA'd while tile i computes; wait TENSORcnt<=2 keeps exactly the
// next tile's two tensor ops in flight (in-order completion per wave).
// ---------------------------------------------------------------------------
__global__ __launch_bounds__(128)
void attn_kernel(const unsigned short* __restrict__ qb,
                 const unsigned short* __restrict__ kbuf,
                 const unsigned short* __restrict__ vtb,
                 unsigned short* __restrict__ ob) {
    __shared__ unsigned short Ks[2][32 * 128];    // [buf][key_local][d]
    __shared__ unsigned short Vts[2][128 * 32];   // [buf][d][key_local]
    __shared__ unsigned short Pl[4][16 * 32];     // per-wave P relayout buffer

    const int tid  = threadIdx.x;
    const int lane = tid & 31;
    const int w    = tid >> 5;
    const int h    = lane >> 4;
    const int l16  = lane & 15;

    const int bh     = blockIdx.y;
    const int qblock = blockIdx.x * 64;
    const int qbase  = qblock + w * 16;

    const unsigned short* Qp = qb   + (size_t)bh * LL * DD;
    const unsigned short* Kp = kbuf + (size_t)bh * LL * DD;
    const unsigned short* Vt = vtb  + (size_t)bh * DD * LL;

    Frag16 qa[4];
#pragma unroll
    for (int kk = 0; kk < 4; ++kk) {
        int row = qbase + l16;
        qa[kk].q[0] = *(const uint4*)&Qp[(size_t)row * DD + kk * 32 + 16 * h];
        qa[kk].q[1] = *(const uint4*)&Qp[(size_t)row * DD + kk * 32 + 16 * h + 8];
    }

    float m_[8], l_[8];
    v8f o[8];
    const v8f vzero = {0.f, 0.f, 0.f, 0.f, 0.f, 0.f, 0.f, 0.f};
#pragma unroll
    for (int v = 0; v < 8; ++v) { m_[v] = -3.0e38f; l_[v] = 0.0f; }
#pragma unroll
    for (int dt = 0; dt < 8; ++dt) o[dt] = vzero;

    const float scale = 0.08838834764831845f;   // 1/sqrt(128)
    const int nkb = (qblock >> 5) + 2;          // covers rows qblock..qblock+63

    // prologue: stage tile 0 into buffer 0
    if (tid < 32) {
        tdm_load_2d_bf16(lds_off(Ks[0]), Kp, DD, LL, DD, 32,
                         (unsigned long long)DD);
        tdm_load_2d_bf16(lds_off(Vts[0]), Vt, LL, DD, 32, DD,
                         (unsigned long long)LL);
    }

    for (int kbi = 0; kbi < nkb; ++kbi) {
        const int k0  = kbi * 32;
        const int cur = kbi & 1;

        if (tid < 32) {
            if (kbi + 1 < nkb) {
                // prefetch next tile into the other buffer (its previous
                // readers finished at the trailing barrier of tile kbi-1)
                const int kn = (kbi + 1) * 32;
                tdm_load_2d_bf16(lds_off(Ks[cur ^ 1]), Kp + (size_t)kn * DD,
                                 DD, LL, DD, 32, (unsigned long long)DD);
                tdm_load_2d_bf16(lds_off(Vts[cur ^ 1]), Vt + kn,
                                 LL, DD, 32, DD, (unsigned long long)LL);
                wait_tensorcnt<2>();   // current tile done; next tile in flight
            } else {
                wait_tensorcnt<0>();
            }
        }
        __syncthreads();

        if (k0 <= qbase + 15) {   // wave-uniform causal skip
            const unsigned short* Kc = Ks[cur];
            const unsigned short* Vc = Vts[cur];

            v8f s0 = vzero, s1 = vzero;
#pragma unroll
            for (int kk = 0; kk < 4; ++kk) {
                Frag16 b0, b1;
                b0.q[0] = *(const uint4*)&Kc[l16 * 128 + kk * 32 + 16 * h];
                b0.q[1] = *(const uint4*)&Kc[l16 * 128 + kk * 32 + 16 * h + 8];
                b1.q[0] = *(const uint4*)&Kc[(16 + l16) * 128 + kk * 32 + 16 * h];
                b1.q[1] = *(const uint4*)&Kc[(16 + l16) * 128 + kk * 32 + 16 * h + 8];
                s0 = wmma_bf16(qa[kk].v, b0.v, s0);
                s1 = wmma_bf16(qa[kk].v, b1.v, s1);
            }

            float p0[8], p1[8];
#pragma unroll
            for (int v = 0; v < 8; ++v) {
                int qrow = qbase + v + 8 * h;
                float x0 = s0[v] * scale;
                float x1 = s1[v] * scale;
                if (k0 + l16 > qrow)      x0 = -3.0e38f;
                if (k0 + 16 + l16 > qrow) x1 = -3.0e38f;
                float mx = fmaxf(x0, x1);
#pragma unroll
                for (int off = 1; off <= 8; off <<= 1)
                    mx = fmaxf(mx, __shfl_xor(mx, off, 32));
                float mnew = fmaxf(m_[v], mx);
                float a  = __expf(m_[v] - mnew);
                float e0 = __expf(x0 - mnew);
                float e1 = __expf(x1 - mnew);
                float rs = e0 + e1;
#pragma unroll
                for (int off = 1; off <= 8; off <<= 1)
                    rs += __shfl_xor(rs, off, 32);
                l_[v] = l_[v] * a + rs;
                m_[v] = mnew;
                p0[v] = e0; p1[v] = e1;
#pragma unroll
                for (int dt = 0; dt < 8; ++dt) o[dt][v] *= a;
            }

            unsigned short* P = &Pl[w][0];
#pragma unroll
            for (int v = 0; v < 8; ++v) {
                int row = v + 8 * h;
                P[row * 32 + l16]      = f32_to_bf16(p0[v]);
                P[row * 32 + 16 + l16] = f32_to_bf16(p1[v]);
            }
            wave_fence();
            Frag16 pa;
            pa.q[0] = *(const uint4*)&P[l16 * 32 + 16 * h];
            pa.q[1] = *(const uint4*)&P[l16 * 32 + 16 * h + 8];

#pragma unroll
            for (int dt = 0; dt < 8; ++dt) {
                Frag16 vb;
                int d = dt * 16 + l16;
                vb.q[0] = *(const uint4*)&Vc[d * 32 + 16 * h];
                vb.q[1] = *(const uint4*)&Vc[d * 32 + 16 * h + 8];
                o[dt] = wmma_bf16(pa.v, vb.v, o[dt]);
            }
        }
        __syncthreads();   // all readers done before this buffer is re-staged
    }

    const int b    = bh >> 4;
    const int head = bh & 15;
    float inv[8];
#pragma unroll
    for (int v = 0; v < 8; ++v) inv[v] = 1.0f / l_[v];
#pragma unroll
    for (int dt = 0; dt < 8; ++dt) {
#pragma unroll
        for (int v = 0; v < 8; ++v) {
            int qrow = qbase + v + 8 * h;
            float val = o[dt][v] * inv[v];
            ob[((size_t)b * LL + qrow) * (NH * DD) + head * DD + dt * 16 + l16] =
                f32_to_bf16(val);
        }
    }
}

// ---------------------------------------------------------------------------
// Kernel 4: output projection.  out = A[8192,2048](bf16) @ Wo[2048,2048]
// A tile double-buffered via TDM (wait TENSORcnt<=1 keeps next tile in
// flight); Wo converted f32->bf16 on VALU.
// ---------------------------------------------------------------------------
__global__ __launch_bounds__(256)
void out_gemm_kernel(const unsigned short* __restrict__ A,
                     const float* __restrict__ Wo,
                     float* __restrict__ out) {
    __shared__ unsigned short As[2][128 * 32];   // [buf][m][k]
    __shared__ unsigned short Bs[128 * 32];      // transposed [n][k]

    const int tid  = threadIdx.x;
    const int lane = tid & 31;
    const int w    = tid >> 5;
    const int wm   = w >> 1;
    const int wn   = w & 1;
    const int h    = lane >> 4;
    const int l16  = lane & 15;

    const int n0 = blockIdx.x * 128;
    const int m0 = blockIdx.y * 128;

    v8f acc[2][4];
    const v8f vzero = {0.f, 0.f, 0.f, 0.f, 0.f, 0.f, 0.f, 0.f};
#pragma unroll
    for (int a = 0; a < 2; ++a)
#pragma unroll
        for (int b = 0; b < 4; ++b) acc[a][b] = vzero;

    // prologue: stage first A tile
    if (tid < 32) {
        tdm_load_2d_bf16(lds_off(As[0]), A + (size_t)m0 * HID,
                         32, MROWS, 32, 128, (unsigned long long)HID);
    }

    for (int k0 = 0; k0 < HID; k0 += 32) {
        const int cur = (k0 >> 5) & 1;

        // Wo tile (32x128 f32) -> bf16 transposed [n][k] on VALU
#pragma unroll
        for (int r = 0; r < 4; ++r) {
            int id  = tid + 256 * r;
            int row = id >> 5;
            int c4  = id & 31;
            float4 wv = *(const float4*)(Wo + (size_t)(k0 + row) * HID + n0 + c4 * 4);
            int n = c4 * 4;
            Bs[(n + 0) * 32 + row] = f32_to_bf16(wv.x);
            Bs[(n + 1) * 32 + row] = f32_to_bf16(wv.y);
            Bs[(n + 2) * 32 + row] = f32_to_bf16(wv.z);
            Bs[(n + 3) * 32 + row] = f32_to_bf16(wv.w);
        }
        if (tid < 32) {
            if (k0 + 32 < HID) {
                tdm_load_2d_bf16(lds_off(As[cur ^ 1]),
                                 A + (size_t)m0 * HID + k0 + 32,
                                 32, MROWS, 32, 128, (unsigned long long)HID);
                wait_tensorcnt<1>();   // current tile done; next in flight
            } else {
                wait_tensorcnt<0>();
            }
        }
        __syncthreads();

        Frag16 afr[2];
#pragma unroll
        for (int ms = 0; ms < 2; ++ms) {
            int row = wm * 32 + ms * 16 + l16;
            afr[ms].q[0] = *(const uint4*)&As[cur][row * 32 + 16 * h];
            afr[ms].q[1] = *(const uint4*)&As[cur][row * 32 + 16 * h + 8];
        }
#pragma unroll
        for (int ns = 0; ns < 4; ++ns) {
            Frag16 bfr;
            int n = wn * 64 + ns * 16 + l16;
            bfr.q[0] = *(const uint4*)&Bs[n * 32 + 16 * h];
            bfr.q[1] = *(const uint4*)&Bs[n * 32 + 16 * h + 8];
#pragma unroll
            for (int ms = 0; ms < 2; ++ms)
                acc[ms][ns] = wmma_bf16(afr[ms].v, bfr.v, acc[ms][ns]);
        }
        __syncthreads();
    }

#pragma unroll
    for (int ms = 0; ms < 2; ++ms) {
#pragma unroll
        for (int ns = 0; ns < 4; ++ns) {
#pragma unroll
            for (int v = 0; v < 8; ++v) {
                int gm = m0 + wm * 32 + ms * 16 + v + 8 * h;
                int gn = n0 + wn * 64 + ns * 16 + l16;
                out[(size_t)gm * HID + gn] = acc[ms][ns][v];
            }
        }
    }
}

// ---------------------------------------------------------------------------
// Launch
// ---------------------------------------------------------------------------
extern "C" void kernel_launch(void* const* d_in, const int* in_sizes, int n_in,
                              void* d_out, int out_size, void* d_ws, size_t ws_size,
                              hipStream_t stream) {
    const float* x    = (const float*)d_in[0];   // [4, 2048, 2048]
    const float* wqkv = (const float*)d_in[1];   // [2048, 6144]
    const float* wo   = (const float*)d_in[2];   // [2048, 2048]
    float* out = (float*)d_out;                  // [4, 2048, 2048]

    const size_t S = (size_t)BH * LL * DD;       // 16,777,216 elems per tensor
    unsigned short* qb  = (unsigned short*)d_ws;
    unsigned short* kbf = qb  + S;
    unsigned short* vtb = kbf + S;
    unsigned short* ob  = vtb + S;               // attention output [B, L, H*D]

    qkv_gemm_kernel<<<dim3(NQKV / 128, MROWS / 128), 256, 0, stream>>>(
        x, wqkv, qb, kbf, vtb);

    {
        int total  = BH * LL * (DD / 2);
        int blocks = total / 256;
        rope_kernel<<<blocks, 256, 0, stream>>>(qb, kbf);
    }

    attn_kernel<<<dim3(LL / 64, BH), 128, 0, stream>>>(qb, kbf, vtb, ob);

    out_gemm_kernel<<<dim3(HID / 128, MROWS / 128), 256, 0, stream>>>(ob, wo, out);
}
